// SelfAttention_43928925503745
// MI455X (gfx1250) — matrix operand order repaired and checked by reading, not compiled, not run on hardware
//
#include <hip/hip_runtime.h>

// ---------------- problem constants ----------------
constexpr int Bn = 4;          // batch
constexpr int Cc = 256;        // channels
constexpr int Hh = 4;          // heads
constexpr int Dd = 64;         // head dim
constexpr int Nn = 4096;       // tokens (64*64)
constexpr int CO = 768;        // 3*C
constexpr size_t MATS = (size_t)Bn * Hh * Nn * Dd;  // == Bn*Nn*Cc = 4,194,304

typedef __attribute__((ext_vector_type(16))) __bf16 v16bf;
typedef __attribute__((ext_vector_type(8)))  float  v8f;
typedef unsigned int u32x4 __attribute__((ext_vector_type(4)));
typedef unsigned int u32x8 __attribute__((ext_vector_type(8)));

// ---------------- WMMA helpers ----------------
__device__ __forceinline__ v8f wmma_bf16(v16bf a, v16bf b, v8f c) {
    return __builtin_amdgcn_wmma_f32_16x16x32_bf16(
        false, a, false, b, (short)0, c, false, false);
}

// A-matrix 16x32 bf16 fragment from row-major f32 source (converted).
__device__ __forceinline__ v16bf load_a_f32(const float* __restrict__ src, int ld,
                                            int row0, int k0, int lane) {
    const float* p = src + (size_t)(row0 + (lane & 15)) * ld + k0 + ((lane >> 4) << 4);
    float4 f0 = *(const float4*)(p);
    float4 f1 = *(const float4*)(p + 4);
    float4 f2 = *(const float4*)(p + 8);
    float4 f3 = *(const float4*)(p + 12);
    v16bf a;
    a[0] = (__bf16)f0.x;  a[1] = (__bf16)f0.y;  a[2]  = (__bf16)f0.z;  a[3]  = (__bf16)f0.w;
    a[4] = (__bf16)f1.x;  a[5] = (__bf16)f1.y;  a[6]  = (__bf16)f1.z;  a[7]  = (__bf16)f1.w;
    a[8] = (__bf16)f2.x;  a[9] = (__bf16)f2.y;  a[10] = (__bf16)f2.z;  a[11] = (__bf16)f2.w;
    a[12] = (__bf16)f3.x; a[13] = (__bf16)f3.y; a[14] = (__bf16)f3.z;  a[15] = (__bf16)f3.w;
    return a;
}

// Same A layout from a bf16 row-major source. Two aligned 16B loads per lane.
__device__ __forceinline__ v16bf load_a_bf16(const __bf16* __restrict__ src, int ld,
                                             int row0, int k0, int lane) {
    const __bf16* p = src + (size_t)(row0 + (lane & 15)) * ld + k0 + ((lane >> 4) << 4);
    union { uint4 u[2]; v16bf v; } c;
    c.u[0] = *(const uint4*)(p);
    c.u[1] = *(const uint4*)(p + 8);
    return c.v;
}

// B-matrix 32x16 bf16 fragment where element(K,N) = src[N][K] (B = src^T).
__device__ __forceinline__ v16bf load_bT(const __bf16* __restrict__ src, int ld,
                                         int n0, int k0, int lane) {
    const __bf16* p = src + (size_t)(n0 + (lane & 15)) * ld + k0 + ((lane >> 4) << 4);
    union { uint4 u[2]; v16bf v; } c;
    c.u[0] = *(const uint4*)(p);
    c.u[1] = *(const uint4*)(p + 8);
    return c.v;
}

// ---------------- Tensor Data Mover (async 2D tile -> LDS) ----------------
// Builds a D# (groups 0+1, 2D tile) in SGPRs and issues TENSOR_LOAD_TO_LDS.
// cfg packs group1[31:0]: data_size=2B (bit16), pad_enable (bit20),
// pad_interval (bits24:22), pad_amount (bits31:25).
// OOB rows/cols (tensor_dim clamped to 0) read as zero -> safe phantom tiles.
__device__ __forceinline__ void tdm_load_2d(const void* gaddr, unsigned lds_off,
                                            unsigned cfg,
                                            unsigned tile_d0, unsigned tile_d1,
                                            unsigned long long stride_d0,
                                            unsigned long long tensor_d0,
                                            unsigned long long tensor_d1) {
    unsigned long long ga = (unsigned long long)gaddr;
    u32x4 g0;
    g0[0] = 1u;                                                  // count=1 (user)
    g0[1] = lds_off;                                             // LDS byte addr
    g0[2] = (unsigned)ga;                                        // global addr lo
    g0[3] = ((unsigned)(ga >> 32) & 0x01FFFFFFu) | 0x80000000u;  // addr hi | type=2
    u32x8 g1;
    g1[0] = cfg;
    g1[1] = (unsigned)(tensor_d0 << 16);                          // dim0[15:0]
    g1[2] = (unsigned)((tensor_d0 >> 16) | (tensor_d1 << 16));    // dim0 hi | dim1 lo
    g1[3] = (unsigned)((tensor_d1 >> 16) | (tile_d0 << 16));      // dim1 hi | tile0
    g1[4] = tile_d1 & 0xffffu;                                    // tile1 (tile2=0)
    g1[5] = (unsigned)stride_d0;                                  // dim0 stride lo
    g1[6] = (unsigned)(stride_d0 >> 32) & 0xffffu;                // stride hi
    g1[7] = 0u;
    asm volatile("tensor_load_to_lds %0, %1" :: "s"(g0), "s"(g1) : "memory");
}
__device__ __forceinline__ unsigned lds_off(const void* p) {
    return (unsigned)(unsigned long long)p;   // flat LDS addr: offset in low 32b
}
// pad cfg: 64-elem rows (32 DW -> interval code 4), +16B pad (amount code 3)
constexpr unsigned TDM_CFG_ROW64 = (1u << 16) | (1u << 20) | (4u << 22) | (3u << 25);
// pad cfg: 32-elem rows (16 DW -> interval code 3), +16B pad (amount code 3)
constexpr unsigned TDM_CFG_ROW32 = (1u << 16) | (1u << 20) | (3u << 22) | (3u << 25);

// =====================================================================
// Kernel 0: transpose + convert  x[b][c][n] f32  ->  xT[b][n][c] bf16
// =====================================================================
__global__ __launch_bounds__(128) void k_xt(const float* __restrict__ x,
                                            __bf16* __restrict__ xT) {
    __shared__ __bf16 lt[64 * 40];   // [n][c] pad 40 (80B rows, 16B aligned)
    const int t = threadIdx.x;
    const int n0 = blockIdx.x * 64;
    const int c0 = blockIdx.y * 32;
    const int b  = blockIdx.z;
    {
        int cl = t >> 2;             // 0..31
        int nl = (t & 3) << 4;       // 0,16,32,48
        const float* g = x + ((size_t)b * Cc + c0 + cl) * Nn + n0 + nl;
        float4 f0 = *(const float4*)(g);
        float4 f1 = *(const float4*)(g + 4);
        float4 f2 = *(const float4*)(g + 8);
        float4 f3 = *(const float4*)(g + 12);
        float v[16] = {f0.x, f0.y, f0.z, f0.w, f1.x, f1.y, f1.z, f1.w,
                       f2.x, f2.y, f2.z, f2.w, f3.x, f3.y, f3.z, f3.w};
#pragma unroll
        for (int j = 0; j < 16; ++j) lt[(nl + j) * 40 + cl] = (__bf16)v[j];
    }
    __syncthreads();
    {
        int nl = t >> 1;             // 0..63
        int ch = (t & 1) << 4;       // 0,16
        const uint4* l = (const uint4*)&lt[nl * 40 + ch];
        uint4* o = (uint4*)(xT + ((size_t)b * Nn + n0 + nl) * Cc + c0 + ch);
        o[0] = l[0];
        o[1] = l[1];
    }
}

// =====================================================================
// Kernel 1: QKV GEMM.  qkv[o,i] = sum_c wqkv[o,c] * xT[b][i][c]
// xT tiles via double-buffered TDM (unconditional issue; last tile is a
// clamped phantom). Q pre-scaled by 1/8. Q,K -> [b][h][n][d]; V -> [b][h][d][n].
// =====================================================================
__global__ __launch_bounds__(128) void k_qkv(const __bf16* __restrict__ xT,
                                             const float* __restrict__ wqkv,
                                             __bf16* __restrict__ qkvst) {
    __shared__ __bf16 lxT[2][64 * 40];  // [i][c] tile, pad 40, double buffered

    const int t = threadIdx.x;
    const int lane = t & 31;
    const int wv = t >> 5;
    const int i0 = blockIdx.x * 64;
    const int o0 = blockIdx.y * 64;
    const int b  = blockIdx.z;

    const __bf16* xrow = xT + (size_t)b * Nn * Cc + (size_t)i0 * Cc;

    if (wv == 0)   // prime: tile cc=0 -> buf 0
        tdm_load_2d(xrow, lds_off(lxT[0]), TDM_CFG_ROW32,
                    32, 64, Cc, Cc, Nn - i0);

    const v8f zero = {0.f, 0.f, 0.f, 0.f, 0.f, 0.f, 0.f, 0.f};
    v8f acc[4] = {zero, zero, zero, zero};

    int buf = 0;
    for (int cc = 0; cc < Cc; cc += 32) {
        __syncthreads();                   // old buf^1 readers done
        if (wv == 0) {                     // unconditional next-tile issue
            tdm_load_2d(xrow + cc + 32, lds_off(lxT[buf ^ 1]), TDM_CFG_ROW32,
                        32, 64, Cc, (unsigned long long)(Cc - cc - 32), Nn - i0);
            __builtin_amdgcn_s_wait_tensorcnt(1);   // current tile done
        }
        __syncthreads();                   // publish current tile

        v16bf a = load_a_f32(wqkv, Cc, o0 + wv * 16, cc, lane);
        const __bf16* lb = lxT[buf];
#pragma unroll
        for (int ct = 0; ct < 4; ++ct)
            acc[ct] = wmma_bf16(a, load_bT(lb, 40, ct * 16, 0, lane), acc[ct]);
        buf ^= 1;
    }

    const int mat = o0 >> 8;            // uniform: 0=Q 1=K 2=V
    if (mat == 2) {                     // V transposed: [b][h][dd][i]
#pragma unroll
        for (int ct = 0; ct < 4; ++ct) {
            int i = i0 + ct * 16 + (lane & 15);
#pragma unroll
            for (int r = 0; r < 8; ++r) {
                int o = o0 + wv * 16 + ((lane >> 4) << 3) + r;
                int oc = o & 255;
                int h = oc >> 6, dd = oc & 63;
                qkvst[2 * MATS + ((size_t)(b * Hh + h) * Dd + dd) * Nn + i] =
                    (__bf16)acc[ct][r];
            }
        }
    } else {
        const float sc = (mat == 0) ? 0.125f : 1.0f;   // fold softmax scale into Q
        const size_t mbase = (size_t)mat * MATS;
#pragma unroll
        for (int ct = 0; ct < 4; ++ct) {
            int i = i0 + ct * 16 + (lane & 15);
#pragma unroll
            for (int r = 0; r < 8; ++r) {
                int o = o0 + wv * 16 + ((lane >> 4) << 3) + r;
                int oc = o & 255;
                int h = oc >> 6, dd = oc & 63;
                qkvst[mbase + ((size_t)(b * Hh + h) * Nn + i) * Dd + dd] =
                    (__bf16)(acc[ct][r] * sc);
            }
        }
    }
}

// =====================================================================
// Kernel 2: flash attention per (b,head). 4 waves x 16 query rows.
// K/V tiles streamed by double-buffered TDM (DMA overlaps WMMA);
// next-tile issue is unconditional (clamped phantom at the end).
// =====================================================================
__global__ __launch_bounds__(128) void k_attn(const __bf16* __restrict__ qkvst,
                                              __bf16* __restrict__ ost) {
    __shared__ __bf16 lK[2][64 * 72];     // [j][dd]
    __shared__ __bf16 lVT[2][64 * 72];    // [dd][j]
    __shared__ __bf16 lP[4][16 * 72];     // per-wave P tile [i][j]

    const int t = threadIdx.x;
    const int lane = t & 31;
    const int wv = t >> 5;
    const int i0 = blockIdx.x * 64;
    const size_t base = (size_t)blockIdx.y * Nn * Dd;   // (b*4+h)

    const __bf16* Q  = qkvst + base;
    const __bf16* K  = qkvst + MATS + base;
    const __bf16* VT = qkvst + 2 * MATS + base;         // [dd][n]

    v16bf qa0 = load_a_bf16(Q, Dd, i0 + wv * 16, 0, lane);
    v16bf qa1 = load_a_bf16(Q, Dd, i0 + wv * 16, 32, lane);

    const v8f zero = {0.f, 0.f, 0.f, 0.f, 0.f, 0.f, 0.f, 0.f};
    v8f oacc[4] = {zero, zero, zero, zero};
    float mrow[8], lrow[8];
#pragma unroll
    for (int r = 0; r < 8; ++r) { mrow[r] = -1e30f; lrow[r] = 0.f; }

    auto issue_tiles = [&](int jb, int bf) {   // jb may be == Nn (phantom)
        unsigned long long rem = (unsigned long long)(Nn - jb);
        tdm_load_2d(K + (size_t)jb * Dd, lds_off(lK[bf]), TDM_CFG_ROW64,
                    64, 64, Dd, Dd, rem);
        tdm_load_2d(VT + jb, lds_off(lVT[bf]), TDM_CFG_ROW64,
                    64, 64, Nn, rem, Dd);
    };
    if (wv == 0) issue_tiles(0, 0);       // prime pipeline

    int buf = 0;
    for (int jb = 0; jb < Nn; jb += 64) {
        __syncthreads();                  // buf^1 readers (iter-1) done
        if (wv == 0) {
            issue_tiles(jb + 64, buf ^ 1);           // overlap next DMA
            __builtin_amdgcn_s_wait_tensorcnt(2);    // current 2 loads done
        }
        __syncthreads();                  // publish current tiles

        const __bf16* bK = lK[buf];
        const __bf16* bV = lVT[buf];

        // S = (Q/8) * K^T : element(K=dd, N=j) = bK[j][dd]  (contiguous)
        v8f s[4];
#pragma unroll
        for (int ct = 0; ct < 4; ++ct) {
            s[ct] = wmma_bf16(qa0, load_bT(bK, 72, ct * 16, 0, lane), zero);
            s[ct] = wmma_bf16(qa1, load_bT(bK, 72, ct * 16, 32, lane), s[ct]);
        }

        // online softmax (C-layout: element r = row 8*(lane/16)+r, col lane%16)
#pragma unroll
        for (int r = 0; r < 8; ++r) {
            float mv = fmaxf(fmaxf(s[0][r], s[1][r]), fmaxf(s[2][r], s[3][r]));
            mv = fmaxf(mv, __shfl_xor(mv, 1, 32));
            mv = fmaxf(mv, __shfl_xor(mv, 2, 32));
            mv = fmaxf(mv, __shfl_xor(mv, 4, 32));
            mv = fmaxf(mv, __shfl_xor(mv, 8, 32));
            float mnew = fmaxf(mrow[r], mv);
            float alpha = __expf(mrow[r] - mnew);
            float rs = 0.f;
#pragma unroll
            for (int ct = 0; ct < 4; ++ct) {
                float pv = __expf(s[ct][r] - mnew);
                s[ct][r] = pv;
                rs += pv;
            }
            rs += __shfl_xor(rs, 1, 32);
            rs += __shfl_xor(rs, 2, 32);
            rs += __shfl_xor(rs, 4, 32);
            rs += __shfl_xor(rs, 8, 32);
            lrow[r] = lrow[r] * alpha + rs;
            mrow[r] = mnew;
#pragma unroll
            for (int dt = 0; dt < 4; ++dt) oacc[dt][r] *= alpha;
        }

        // P (C-layout f32) -> LDS bf16 (scatter stores; re-read as b128 A-frags)
        __bf16* lp = lP[wv];
#pragma unroll
        for (int ct = 0; ct < 4; ++ct)
#pragma unroll
            for (int r = 0; r < 8; ++r)
                lp[(((lane >> 4) << 3) + r) * 72 + ct * 16 + (lane & 15)] =
                    (__bf16)s[ct][r];
        __syncthreads();

        // O += P * V : element(K=j, N=dd) = bV[dd][j]  (contiguous)
        v16bf pa0 = load_a_bf16(lp, 72, 0, 0, lane);
        v16bf pa1 = load_a_bf16(lp, 72, 0, 32, lane);
#pragma unroll
        for (int dt = 0; dt < 4; ++dt) {
            oacc[dt] = wmma_bf16(pa0, load_bT(bV, 72, dt * 16, 0, lane), oacc[dt]);
            oacc[dt] = wmma_bf16(pa1, load_bT(bV, 72, dt * 16, 32, lane), oacc[dt]);
        }
        buf ^= 1;
    }

    // finalize: divide by row sum, store bf16 O as [b][h][i][dd]
#pragma unroll
    for (int r = 0; r < 8; ++r) {
        float inv = 1.0f / lrow[r];
        int i = i0 + wv * 16 + ((lane >> 4) << 3) + r;
#pragma unroll
        for (int dt = 0; dt < 4; ++dt) {
            int dd = dt * 16 + (lane & 15);
            ost[base + (size_t)i * Dd + dd] = (__bf16)(oacc[dt][r] * inv);
        }
    }
}

// =====================================================================
// Kernel 3: output projection + residual.
// out[b,o,i] = sum_c wproj[o,c] * O[b, c/64, i, c%64] + x[b,o,i]
// =====================================================================
__global__ __launch_bounds__(128) void k_proj(const __bf16* __restrict__ ost,
                                              const float* __restrict__ wproj,
                                              const float* __restrict__ x,
                                              float* __restrict__ out) {
    const int t = threadIdx.x;
    const int lane = t & 31;
    const int wv = t >> 5;
    const int i0 = blockIdx.x * 64;
    const int o0 = blockIdx.y * 64;
    const int b  = blockIdx.z;

    const v8f zero = {0.f, 0.f, 0.f, 0.f, 0.f, 0.f, 0.f, 0.f};
    v8f acc[4] = {zero, zero, zero, zero};

    for (int cc = 0; cc < Cc; cc += 32) {
        v16bf a = load_a_f32(wproj, Cc, o0 + wv * 16, cc, lane);
        const int h  = cc >> 6;
        const int dr = cc & 63;
        const __bf16* ob = ost + (size_t)(b * Hh + h) * Nn * Dd;
#pragma unroll
        for (int ct = 0; ct < 4; ++ct) {
            const __bf16* p = ob + (size_t)(i0 + ct * 16 + (lane & 15)) * Dd +
                              dr + ((lane >> 4) << 4);
            __builtin_prefetch(p + 32, 0, 3);
            union { uint4 u[2]; v16bf v; } c;
            c.u[0] = *(const uint4*)(p);
            c.u[1] = *(const uint4*)(p + 8);
            acc[ct] = wmma_bf16(a, c.v, acc[ct]);
        }
    }

#pragma unroll
    for (int ct = 0; ct < 4; ++ct) {
        int i = i0 + ct * 16 + (lane & 15);
#pragma unroll
        for (int r = 0; r < 8; ++r) {
            int o = o0 + wv * 16 + ((lane >> 4) << 3) + r;
            size_t idx = ((size_t)b * Cc + o) * (size_t)Nn + i;
            out[idx] = acc[ct][r] + x[idx];
        }
    }
}

// =====================================================================
extern "C" void kernel_launch(void* const* d_in, const int* in_sizes, int n_in,
                              void* d_out, int out_size, void* d_ws, size_t ws_size,
                              hipStream_t stream) {
    const float* x     = (const float*)d_in[0];
    const float* wqkv  = (const float*)d_in[1];
    const float* wproj = (const float*)d_in[2];
    float* out = (float*)d_out;

    __bf16* xT    = (__bf16*)d_ws;              // MATS bf16
    __bf16* qkvst = xT + MATS;                  // 3*MATS bf16
    __bf16* ost   = qkvst + 3 * MATS;           // MATS bf16  (total ~42 MB)

    dim3 blk(128);
    k_xt  <<<dim3(Nn / 64, Cc / 32, Bn), blk, 0, stream>>>(x, xT);
    k_qkv <<<dim3(Nn / 64, CO / 64, Bn), blk, 0, stream>>>(xT, wqkv, qkvst);
    k_attn<<<dim3(Nn / 64, Bn * Hh),     blk, 0, stream>>>(qkvst, ost);
    k_proj<<<dim3(Nn / 64, Cc / 64, Bn), blk, 0, stream>>>(ost, wproj, x, out);
}
